// Attention_27960237097134
// MI455X (gfx1250) — matrix operand order repaired and checked by reading, not compiled, not run on hardware
//
#include <hip/hip_runtime.h>

// Problem constants (from reference)
#define B_ 32
#define T_ 2048
#define F_ 2048
#define H_ 1024
#define A_ 512
#define TSPLIT 16

typedef __attribute__((ext_vector_type(16))) __bf16  v16bf;
typedef __attribute__((ext_vector_type(8)))  float   v8f;
typedef __attribute__((ext_vector_type(4)))  float   f32x4;
typedef __attribute__((ext_vector_type(4)))  unsigned int u32x4;
typedef __attribute__((ext_vector_type(2)))  unsigned int u32x2;

union AFrag { v16bf v; unsigned int u[8]; };
union BFrag { v16bf v; u32x4 q[2]; };

// Pack two fp32 -> packed bf16x2 in one dword.
// Preferred: hardware v_cvt_pk_bf16_f32. Fallback: round-nearest bias + v_perm_b32.
__device__ __forceinline__ unsigned int pack2bf(float lo, float hi) {
#if __has_builtin(__builtin_amdgcn_cvt_pk_bf16_f32)
  auto r = __builtin_amdgcn_cvt_pk_bf16_f32(lo, hi);
  if constexpr (sizeof(r) == 4) {
    return __builtin_bit_cast(unsigned int, r);
  }
#endif
  unsigned int a = __float_as_uint(lo) + 0x8000u;   // round-to-nearest
  unsigned int b = __float_as_uint(hi) + 0x8000u;
  // dst = { b[3], b[2], a[3], a[2] }  (b = S0 = concat bytes 7:4)
  return __builtin_amdgcn_perm(b, a, 0x07060302u);  // v_perm_b32
}

// Branch-free tanh: accurate to ~1 ulp of v_exp_f32, saturates correctly.
__device__ __forceinline__ float fast_tanh(float x) {
  float xc = fminf(fmaxf(x, -15.f), 15.f);
  float e2 = __expf(2.f * xc);              // v_exp_f32
  return 1.f - 2.f / (e2 + 1.f);            // v_rcp_f32
}

// ---------------------------------------------------------------------------
// (a) Convert W_a [A,F] fp32 -> bf16 (row-major: each row n is a B-matrix
//     column, contiguous in K=f -> plain 128-bit B-fragment loads).
// ---------------------------------------------------------------------------
__global__ __launch_bounds__(256) void cvt_wa_kernel(const float* __restrict__ wa,
                                                     unsigned int* __restrict__ out2) {
  int i = blockIdx.x * 256 + threadIdx.x;          // one thread per 4 elements
  f32x4 f = ((const f32x4*)wa)[i];
  u32x2 o;
  o.x = pack2bf(f.x, f.y);
  o.y = pack2bf(f.z, f.w);
  ((u32x2*)out2)[i] = o;
}

// ---------------------------------------------------------------------------
// (b) hp[b,a] = hidden[b,:]·W_h[a,:] + b_h[a] + b_a[a]   (both biases folded)
// ---------------------------------------------------------------------------
__global__ __launch_bounds__(256) void hp_kernel(const float* __restrict__ hidden,
                                                 const float* __restrict__ Wh,
                                                 const float* __restrict__ bh,
                                                 const float* __restrict__ ba,
                                                 float* __restrict__ hp) {
  int idx = blockIdx.x * 256 + threadIdx.x;        // 0 .. B*A-1
  int b = idx / A_, a = idx % A_;
  const f32x4* h4 = (const f32x4*)(hidden + (size_t)b * H_);
  const f32x4* w4 = (const f32x4*)(Wh + (size_t)a * H_);
  float s = 0.f;
  #pragma unroll 4
  for (int k = 0; k < H_ / 4; ++k) {
    f32x4 x = h4[k], y = w4[k];
    s += x.x * y.x + x.y * y.y + x.z * y.z + x.w * y.w;
  }
  hp[idx] = s + bh[a] + ba[a];
}

// ---------------------------------------------------------------------------
// (c) Fused GEMM + energy.
//     Workgroup: 8 waves, covers 32 t-rows (2 M-tiles) x all N=512.
//     Wave (mt = w>>2, ng = w&3): one 16-row M-tile, 8 N-tiles (n in [128*ng,+128)).
//     K-loop over F in steps of 32 with v_wmma_f32_16x16x32_bf16.
//     Epilogue: e[t] += v_w[n]*tanh(acc + hp[b,n]); lane/LDS reduce; *1/sqrt(A).
// ---------------------------------------------------------------------------
__global__ __launch_bounds__(256, 1) void gemm_energy_kernel(
    const float* __restrict__ feat, const unsigned short* __restrict__ wa_bf,
    const float* __restrict__ hp, const float* __restrict__ vw,
    const float* __restrict__ vb, float* __restrict__ energy) {
  const int tid   = threadIdx.x;
  const int wave  = tid >> 5;
  const int lane  = tid & 31;
  const int lrow  = lane & 15;
  const int lhalf = lane >> 4;
  const int b     = blockIdx.y;
  const int tbase = blockIdx.x * 32;
  const int mt    = wave >> 2;          // 0..1 : which 16-row tile
  const int nbase = (wave & 3) * 128;   // 8 n-tiles of 16 per wave

  __shared__ float e_lds[32];
  if (tid < 32) e_lds[tid] = 0.f;
  __syncthreads();

  v8f acc[8] = {};

  const float* aptr =
      feat + ((size_t)b * T_ + (tbase + mt * 16 + lrow)) * F_ + lhalf * 8;
  const unsigned short* brow = wa_bf + (size_t)(nbase + lrow) * F_ + lhalf * 8;

  for (int kk = 0; kk < F_; kk += 32) {
    __builtin_prefetch((const void*)(aptr + kk + 512), 0, 0);  // global_prefetch_b8

    // A fragment: rows of `features`, fp32 -> bf16 on the fly.
    // Layout: elems 0..7 = K kb..kb+7, elems 8..15 = K kb+16..kb+23, kb=8*lhalf.
    f32x4 f0 = ((const f32x4*)(aptr + kk))[0];
    f32x4 f1 = ((const f32x4*)(aptr + kk))[1];
    f32x4 f2 = ((const f32x4*)(aptr + kk + 16))[0];
    f32x4 f3 = ((const f32x4*)(aptr + kk + 16))[1];
    AFrag af;
    af.u[0] = pack2bf(f0.x, f0.y); af.u[1] = pack2bf(f0.z, f0.w);
    af.u[2] = pack2bf(f1.x, f1.y); af.u[3] = pack2bf(f1.z, f1.w);
    af.u[4] = pack2bf(f2.x, f2.y); af.u[5] = pack2bf(f2.z, f2.w);
    af.u[6] = pack2bf(f3.x, f3.y); af.u[7] = pack2bf(f3.z, f3.w);

    #pragma unroll
    for (int j = 0; j < 8; ++j) {
      // B fragment: column n = nbase+16j+lrow of a_proj = row of W_a (bf16),
      // K: elems 0..7 at kk + 8*lhalf, elems 8..15 at kk + 16 + 8*lhalf.
      BFrag bfr;
      const u32x4* bp = (const u32x4*)(brow + (size_t)j * 16 * F_ + kk);
      bfr.q[0] = bp[0];
      bfr.q[1] = bp[2];
      acc[j] = __builtin_amdgcn_wmma_f32_16x16x32_bf16(
          false, af.v, false, bfr.v, (short)0, acc[j], false, false);
    }
  }

  // Epilogue: esum[r] over this wave's 128 n-values, row m = mtbase + r + 8*lhalf
  float esum[8];
  #pragma unroll
  for (int r = 0; r < 8; ++r) esum[r] = 0.f;
  #pragma unroll
  for (int j = 0; j < 8; ++j) {
    int n   = nbase + j * 16 + lrow;
    float h = hp[b * A_ + n];
    float v = vw[n];
    #pragma unroll
    for (int r = 0; r < 8; ++r) esum[r] += v * fast_tanh(acc[j][r] + h);
  }
  // Reduce over the 16 lanes that share a row (xor masks stay within a half).
  #pragma unroll
  for (int r = 0; r < 8; ++r) {
    float s = esum[r];
    s += __shfl_xor(s, 1, 32);
    s += __shfl_xor(s, 2, 32);
    s += __shfl_xor(s, 4, 32);
    s += __shfl_xor(s, 8, 32);
    if (lrow == 0) atomicAdd(&e_lds[mt * 16 + lhalf * 8 + r], s);  // ds_add_f32
  }
  __syncthreads();
  if (tid < 32) {
    float e = (e_lds[tid] + vb[0]) * 0.04419417382415922f;  // 1/sqrt(512)
    energy[(size_t)b * T_ + tbase + tid] = e;
  }
}

// ---------------------------------------------------------------------------
// (d) Softmax over T per batch. One workgroup per b, 8 elems/thread.
// ---------------------------------------------------------------------------
__global__ __launch_bounds__(256) void softmax_kernel(const float* __restrict__ energy,
                                                      float* __restrict__ alpha) {
  int b = blockIdx.x, tid = threadIdx.x;
  __shared__ float red[256];
  float vals[8];
  float m = -3.402823466e38f;
  #pragma unroll
  for (int i = 0; i < 8; ++i) {
    vals[i] = energy[(size_t)b * T_ + tid + i * 256];
    m = fmaxf(m, vals[i]);
  }
  red[tid] = m; __syncthreads();
  for (int off = 128; off > 0; off >>= 1) {
    if (tid < off) red[tid] = fmaxf(red[tid], red[tid + off]);
    __syncthreads();
  }
  float M = red[0]; __syncthreads();
  float s = 0.f;
  #pragma unroll
  for (int i = 0; i < 8; ++i) { vals[i] = __expf(vals[i] - M); s += vals[i]; }
  red[tid] = s; __syncthreads();
  for (int off = 128; off > 0; off >>= 1) {
    if (tid < off) red[tid] += red[tid + off];
    __syncthreads();
  }
  float inv = 1.f / red[0];
  #pragma unroll
  for (int i = 0; i < 8; ++i)
    alpha[(size_t)b * T_ + tid + i * 256] = vals[i] * inv;
}

// ---------------------------------------------------------------------------
// (e) Partial context: grid (F/256, B, TSPLIT); each WG sums 128 t-rows.
//     One prefetch per 8-row group, FMA loop otherwise pure global_load_b32.
// ---------------------------------------------------------------------------
__global__ __launch_bounds__(256) void ctx_partial_kernel(
    const float* __restrict__ feat, const float* __restrict__ alpha,
    float* __restrict__ partial) {
  int f = blockIdx.x * 256 + threadIdx.x;
  int b = blockIdx.y, s = blockIdx.z;
  __shared__ float a_lds[128];
  if (threadIdx.x < 128) a_lds[threadIdx.x] = alpha[(size_t)b * T_ + s * 128 + threadIdx.x];
  __syncthreads();
  const float* fp = feat + ((size_t)b * T_ + (size_t)s * 128) * F_ + f;
  float acc = 0.f;
  for (int t8 = 0; t8 < 128; t8 += 8) {
    __builtin_prefetch((const void*)(fp + (size_t)(t8 + 16) * F_), 0, 0);
    #pragma unroll
    for (int u = 0; u < 8; ++u) {
      int t = t8 + u;
      acc += a_lds[t] * fp[(size_t)t * F_];
    }
  }
  partial[((size_t)b * TSPLIT + s) * F_ + f] = acc;
}

// (f) Reduce TSPLIT partials -> context.
__global__ __launch_bounds__(256) void ctx_reduce_kernel(const float* __restrict__ partial,
                                                         float* __restrict__ ctx) {
  int idx = blockIdx.x * 256 + threadIdx.x;  // 0 .. B*F-1
  int b = idx / F_, f = idx % F_;
  float s = 0.f;
  #pragma unroll
  for (int i = 0; i < TSPLIT; ++i) s += partial[((size_t)b * TSPLIT + i) * F_ + f];
  ctx[idx] = s;
}

// ---------------------------------------------------------------------------
extern "C" void kernel_launch(void* const* d_in, const int* in_sizes, int n_in,
                              void* d_out, int out_size, void* d_ws, size_t ws_size,
                              hipStream_t stream) {
  const float* feat   = (const float*)d_in[0];  // [B,T,F]
  const float* hidden = (const float*)d_in[1];  // [B,H]
  const float* Wh     = (const float*)d_in[2];  // [A,H]
  const float* bh     = (const float*)d_in[3];  // [A]
  const float* Wa     = (const float*)d_in[4];  // [A,F]
  const float* ba     = (const float*)d_in[5];  // [A]
  const float* vw     = (const float*)d_in[6];  // [1,A]
  const float* vb     = (const float*)d_in[7];  // [1]

  float* ctx_out   = (float*)d_out;                     // [B,F]
  float* alpha_out = (float*)d_out + (size_t)B_ * F_;   // [B,T]

  // Workspace layout (bytes): hp | energy | partials | W_a(bf16)  (~6.6 MB)
  char* ws = (char*)d_ws;
  float* hp            = (float*)(ws);                        //  65536 B
  float* energy        = (float*)(ws + 65536);                // 262144 B
  float* partial       = (float*)(ws + 327680);               // 4 MiB
  unsigned int* wa_bf2 = (unsigned int*)(ws + 4521984);       // 2 MiB

  cvt_wa_kernel<<<(A_ * F_ / 4) / 256, 256, 0, stream>>>(Wa, wa_bf2);
  hp_kernel<<<(B_ * A_) / 256, 256, 0, stream>>>(hidden, Wh, bh, ba, hp);
  gemm_energy_kernel<<<dim3(T_ / 32, B_), 256, 0, stream>>>(
      feat, (const unsigned short*)wa_bf2, hp, vw, vb, energy);
  softmax_kernel<<<B_, 256, 0, stream>>>(energy, alpha_out);
  ctx_partial_kernel<<<dim3(F_ / 256, B_, TSPLIT), 256, 0, stream>>>(
      feat, alpha_out, partial);
  ctx_reduce_kernel<<<(B_ * F_) / 256, 256, 0, stream>>>(partial, ctx_out);
}
